// TweetMamba_59631325938126
// MI455X (gfx1250) — compile-verified, matrix operand order checked by reading, not run on hardware
//
#include <hip/hip_runtime.h>
#include <hip/hip_bf16.h>
#include <math.h>

// ---------------------------------------------------------------------------
// Problem constants (from reference)
// ---------------------------------------------------------------------------
#define BB   4
#define TT   1000
#define WW   32
#define DD   200
#define LL   1001          // T + 1 (cls inserted)
#define POS  500           // T//2
#define LH   501           // scan length needed per branch (0..pos)
#define ED   400
#define NN   16
#define DTR  13
#define DBCS 48            // padded stride for dbc rows (45 -> 48)
#define MTOT (BB * LH)     // 2004 rows per branch GEMM

// Padded-K / tile constants for the three WMMA GEMMs
#define KP1 224            // in_proj:  K=200 -> 224, 7 k-steps
#define KT1 7
#define NT1 25             // N=400 -> 25 n-tiles
#define KP2 416            // xproj:    K=400 -> 416, 13 k-steps
#define KT2 13
#define NT2 3              // N=45 -> 3 n-tiles
#define KP3 32             // dt:       K=13 -> 32, 1 k-step
#define KT3 1
#define NT3 25             // N=400 -> 25 n-tiles

typedef __attribute__((ext_vector_type(16))) __bf16 v16bf;
typedef __attribute__((ext_vector_type(8)))  __bf16 v8bf;
typedef __attribute__((ext_vector_type(8)))  float  v8f;

// ---------------------------------------------------------------------------
// WMMA tile core: acc += A(16 x 32*KT bf16, row-major padded) * B(swizzled)
// A per ISA 7.12.2: lane%16 = M; lane needs arow[kt*32 + 8*(lane/16) + 0..7]
// and arow[kt*32 + 16 + 8*(lane/16) + 0..7] -> two 16B contiguous loads.
// B pre-swizzled so each lane reads 32 contiguous bytes per k-step.
// Uniform control flow: EXEC all-ones around every v_wmma.
// ---------------------------------------------------------------------------
template <int KT>
__device__ __forceinline__ v8f gemm_tile(const __bf16* __restrict__ arow,
                                         const __bf16* __restrict__ btile) {
  const int lane = threadIdx.x & 31;
  const int half = lane >> 4;
  v8f acc = {};
#pragma unroll
  for (int kt = 0; kt < KT; ++kt) {
    const __bf16* ap = arow + kt * 32 + half * 8;
    const v8bf alo = *(const v8bf*)(ap);
    const v8bf ahi = *(const v8bf*)(ap + 16);
    const __bf16* bp = btile + ((size_t)kt * 32 + lane) * 16;
    const v8bf blo = *(const v8bf*)(bp);
    const v8bf bhi = *(const v8bf*)(bp + 8);
    const v16bf av = __builtin_shufflevector(alo, ahi, 0, 1, 2, 3, 4, 5, 6, 7,
                                             8, 9, 10, 11, 12, 13, 14, 15);
    const v16bf bv = __builtin_shufflevector(blo, bhi, 0, 1, 2, 3, 4, 5, 6, 7,
                                             8, 9, 10, 11, 12, 13, 14, 15);
    acc = __builtin_amdgcn_wmma_f32_16x16x32_bf16(false, av, false, bv,
                                                  (short)0, acc, false, false);
  }
  return acc;
}

// ---------------------------------------------------------------------------
// K1: word-attention pooling + cls insert + RMSNorm -> xn[B][L][D]
// ---------------------------------------------------------------------------
__global__ void pool_norm_kernel(const float* __restrict__ ids,
                                 const int* __restrict__ ntw,
                                 const int* __restrict__ nwd,
                                 const float* __restrict__ cls,
                                 const float* __restrict__ wat,
                                 const float* __restrict__ bat,
                                 const float* __restrict__ nw,
                                 float* __restrict__ xn) {
  const int b = blockIdx.x / LL;
  const int l = blockIdx.x % LL;
  const int tid = threadIdx.x;

  __shared__ float row[DD];
  __shared__ float red[256];
  __shared__ float attn[WW];

  if (l == POS) {
    for (int d = tid; d < DD; d += blockDim.x) row[d] = cls[d];
    __syncthreads();
  } else {
    const int t = (l < POS) ? l : (l - 1);
    const float* base = ids + ((size_t)b * TT + t) * WW * DD;
    {
      const int w = tid >> 3, sub = tid & 7;
      float p = 0.f;
      for (int d = sub; d < DD; d += 8) p += base[w * DD + d] * wat[d];
      red[tid] = p;
    }
    __syncthreads();
    if (tid < WW) {
      float s = 0.f;
      for (int k = 0; k < 8; ++k) s += red[tid * 8 + k];
      s += bat[0];
      attn[tid] = (tid < nwd[b * TT + t]) ? s : -1e30f;
    }
    __syncthreads();
    if (tid < 32) {  // wave 0: masked softmax over 32 words
      float s = attn[tid];
      float mx = s;
      for (int m = 16; m >= 1; m >>= 1) mx = fmaxf(mx, __shfl_xor(mx, m));
      float e = expf(s - mx);
      float sum = e;
      for (int m = 16; m >= 1; m >>= 1) sum += __shfl_xor(sum, m);
      attn[tid] = e / sum;
    }
    __syncthreads();
    const bool keep = (t < ntw[b]);
    for (int d = tid; d < DD; d += blockDim.x) {
      float acc = 0.f;
      for (int w = 0; w < WW; ++w) acc += attn[w] * base[w * DD + d];
      row[d] = keep ? acc : 0.f;
    }
    __syncthreads();
  }

  float ss = 0.f;
  for (int d = tid; d < DD; d += blockDim.x) ss += row[d] * row[d];
  red[tid] = ss;
  __syncthreads();
  for (int s = 128; s > 0; s >>= 1) {
    if (tid < s) red[tid] += red[tid + s];
    __syncthreads();
  }
  const float rs = rsqrtf(red[0] / (float)DD + 1e-5f);
  float* orow = xn + ((size_t)b * LL + l) * DD;
  for (int d = tid; d < DD; d += blockDim.x) orow[d] = row[d] * rs * nw[d];
}

// ---------------------------------------------------------------------------
// Pack xn -> bf16 A1[branch][r][KP1] (rows pre-flipped for backward branch)
// ---------------------------------------------------------------------------
__global__ void pack_a1_kernel(const float* __restrict__ xn,
                               __bf16* __restrict__ a1) {
  const int idx = blockIdx.x * blockDim.x + threadIdx.x;
  if (idx >= 2 * MTOT * KP1) return;
  const int branch = idx / (MTOT * KP1);
  const int rem = idx % (MTOT * KP1);
  const int r = rem / KP1, kp = rem % KP1;
  const int b = r / LH, l = r % LH;
  const int src = branch ? (LL - 1 - l) : l;
  const float v = (kp < DD) ? xn[((size_t)b * LL + src) * DD + kp] : 0.f;
  a1[idx] = (__bf16)v;
}

// ---------------------------------------------------------------------------
// Pack a KxN f32 weight (ldw row stride) into swizzled bf16 WMMA-B operand:
// dst[((ntile*ktcnt + kt)*32 + lane)*16 + 2j(+1)] = w[kb][ntile*16+lane%16]
// with kb = kt*32 + 16*(j/4) + 8*(lane/16) + 2*(j%4). Zero-padded.
// ---------------------------------------------------------------------------
__global__ void pack_w_kernel(const float* __restrict__ src, int ldw, int ncols,
                              int K, int ktcnt, int ntiles,
                              __bf16* __restrict__ dst) {
  const int idx = blockIdx.x * blockDim.x + threadIdx.x;
  if (idx >= ntiles * ktcnt * 32) return;
  const int lane = idx & 31;
  const int t = idx >> 5;
  const int kt = t % ktcnt;
  const int half = lane >> 4;
  const int n = (t / ktcnt) * 16 + (lane & 15);
  __bf16* d = dst + (size_t)idx * 16;
#pragma unroll
  for (int j = 0; j < 8; ++j) {
    const int kb = kt * 32 + ((j >> 2) << 4) + (half << 3) + ((j & 3) << 1);
    const float f0 = (n < ncols && kb < K) ? src[(size_t)kb * ldw + n] : 0.f;
    const float f1 = (n < ncols && kb + 1 < K) ? src[(size_t)(kb + 1) * ldw + n] : 0.f;
    d[2 * j] = (__bf16)f0;
    d[2 * j + 1] = (__bf16)f1;
  }
}

// ---------------------------------------------------------------------------
// K2: u_d = A1_d @ in_proj_d[:, :ED]  (M=2004, Kp=224, N=400) via WMMA
// ---------------------------------------------------------------------------
__global__ void inproj_kernel(const __bf16* __restrict__ a1,
                              const __bf16* __restrict__ w1,
                              float* __restrict__ uf, float* __restrict__ ub) {
  const int branch = blockIdx.z;
  float* u = branch ? ub : uf;
  const __bf16* A = a1 + (size_t)branch * MTOT * KP1;
  const __bf16* W = w1 + (size_t)branch * NT1 * KT1 * 512;

  const int wave = threadIdx.x >> 5;
  const int lane = threadIdx.x & 31;
  const int half = lane >> 4;
  const int ntile = blockIdx.y * 4 + wave;
  if (ntile >= NT1) return;  // wave-uniform
  const int m0 = blockIdx.x * 16;
  const int r = m0 + (lane & 15);
  const int rc = (r < MTOT) ? r : 0;

  const v8f acc = gemm_tile<KT1>(A + (size_t)rc * KP1,
                                 W + (size_t)ntile * KT1 * 512);

  const int n = ntile * 16 + (lane & 15);
#pragma unroll
  for (int i = 0; i < 8; ++i) {
    const int r2 = m0 + i + half * 8;
    if (r2 < MTOT) u[(size_t)r2 * ED + n] = acc[i];
  }
}

// ---------------------------------------------------------------------------
// K3: xp = silu(causal_conv(u) + conv_b); also emits bf16 A2 (K padded 416)
// ---------------------------------------------------------------------------
__global__ void conv_silu_kernel(const float* __restrict__ uf,
                                 const float* __restrict__ ub,
                                 const float* __restrict__ cwf,
                                 const float* __restrict__ cbf,
                                 const float* __restrict__ cwb,
                                 const float* __restrict__ cbb,
                                 float* __restrict__ xpf,
                                 float* __restrict__ xpb,
                                 __bf16* __restrict__ a2) {
  const int branch = blockIdx.z;
  const float* u  = branch ? ub  : uf;
  const float* cw = branch ? cwb : cwf;
  const float* cb = branch ? cbb : cbf;
  float* xp = branch ? xpb : xpf;

  const int idx = blockIdx.x * blockDim.x + threadIdx.x;
  if (idx >= MTOT * KP2) return;
  const int e = idx % KP2;
  const int rl = idx / KP2;
  __bf16* arow2 = a2 + ((size_t)branch * MTOT + rl) * KP2;
  if (e >= ED) {  // zero K-padding
    arow2[e] = (__bf16)0.f;
    return;
  }
  const int b = rl / LH, l = rl % LH;
  float s = cb[e];
#pragma unroll
  for (int k = 0; k < 4; ++k) {
    const int ls = l + k - 3;
    if (ls >= 0) s += u[((size_t)b * LH + ls) * ED + e] * cw[e * 4 + k];
  }
  const float sv = s / (1.f + expf(-s));  // silu
  xp[(size_t)rl * ED + e] = sv;
  arow2[e] = (__bf16)sv;
}

// ---------------------------------------------------------------------------
// K4: dbc = xp @ xproj (M=2004, Kp=416, N=45); also emits bf16 delta A3
// ---------------------------------------------------------------------------
__global__ void xproj_kernel(const __bf16* __restrict__ a2,
                             const __bf16* __restrict__ w2,
                             float* __restrict__ dbcf,
                             float* __restrict__ dbcb,
                             __bf16* __restrict__ a3) {
  const int branch = blockIdx.z;
  float* dbc = branch ? dbcb : dbcf;
  const __bf16* A = a2 + (size_t)branch * MTOT * KP2;
  const __bf16* W = w2 + (size_t)branch * NT2 * KT2 * 512;

  const int wave = threadIdx.x >> 5;
  const int lane = threadIdx.x & 31;
  const int half = lane >> 4;
  if (wave >= NT2) return;  // wave-uniform
  const int m0 = blockIdx.x * 16;
  const int r = m0 + (lane & 15);
  const int rc = (r < MTOT) ? r : 0;

  const v8f acc = gemm_tile<KT2>(A + (size_t)rc * KP2,
                                 W + (size_t)wave * KT2 * 512);

  const int n = wave * 16 + (lane & 15);
#pragma unroll
  for (int i = 0; i < 8; ++i) {
    const int r2 = m0 + i + half * 8;
    if (r2 < MTOT) {
      if (n < DTR + 2 * NN) dbc[(size_t)r2 * DBCS + n] = acc[i];
      if (n < KP3)  // delta (cols < 13), zero-padded to 32
        a3[((size_t)branch * MTOT + r2) * KP3 + n] =
            (__bf16)((n < DTR) ? acc[i] : 0.f);
    }
  }
}

// ---------------------------------------------------------------------------
// K5: dt = softplus(delta @ dtw + dtb)  (M=2004, Kp=32, N=400) via WMMA
// ---------------------------------------------------------------------------
__global__ void dt_kernel(const __bf16* __restrict__ a3,
                          const __bf16* __restrict__ w3,
                          const float* __restrict__ dtbf,
                          const float* __restrict__ dtbb,
                          float* __restrict__ dtf,
                          float* __restrict__ dtb_o) {
  const int branch = blockIdx.z;
  const float* bia = branch ? dtbb : dtbf;
  float* dt = branch ? dtb_o : dtf;
  const __bf16* A = a3 + (size_t)branch * MTOT * KP3;
  const __bf16* W = w3 + (size_t)branch * NT3 * KT3 * 512;

  const int wave = threadIdx.x >> 5;
  const int lane = threadIdx.x & 31;
  const int half = lane >> 4;
  const int ntile = blockIdx.y * 4 + wave;
  if (ntile >= NT3) return;  // wave-uniform
  const int m0 = blockIdx.x * 16;
  const int r = m0 + (lane & 15);
  const int rc = (r < MTOT) ? r : 0;

  const v8f acc = gemm_tile<KT3>(A + (size_t)rc * KP3,
                                 W + (size_t)ntile * KT3 * 512);

  const int n = ntile * 16 + (lane & 15);
  const float bv = bia[n];
#pragma unroll
  for (int i = 0; i < 8; ++i) {
    const int r2 = m0 + i + half * 8;
    if (r2 < MTOT) {
      const float v = acc[i] + bv;
      dt[(size_t)r2 * ED + n] = (v > 20.f) ? v : log1pf(expf(v));
    }
  }
}

// ---------------------------------------------------------------------------
// K6: sequential SSM scan. One lane per (b, e, n); only y[pos] materialized.
// ---------------------------------------------------------------------------
__global__ void scan_kernel(const float* __restrict__ xpf,
                            const float* __restrict__ xpb,
                            const float* __restrict__ dtf,
                            const float* __restrict__ dtb_i,
                            const float* __restrict__ dbcf,
                            const float* __restrict__ dbcb,
                            const float* __restrict__ Alf,
                            const float* __restrict__ Alb,
                            const float* __restrict__ Dpf,
                            const float* __restrict__ Dpb,
                            float* __restrict__ ypos) {
  const int branch = blockIdx.z;
  const float* xp  = branch ? xpb   : xpf;
  const float* dt  = branch ? dtb_i : dtf;
  const float* dbc = branch ? dbcb  : dbcf;
  const float* Al  = branch ? Alb   : Alf;
  const float* Dp  = branch ? Dpb   : Dpf;

  const int n    = threadIdx.x & 15;
  const int eloc = threadIdx.x >> 4;
  const int b    = blockIdx.x / (ED / 16);
  const int e    = (blockIdx.x % (ED / 16)) * 16 + eloc;

  const float Aen = -expf(Al[e * NN + n]);
  float h = 0.f;
  const size_t rbase = (size_t)b * LH;
  for (int l = 0; l < LH; ++l) {
    const size_t r = rbase + l;
    if (l + 8 < LH) {  // -> global_prefetch_b8
      __builtin_prefetch(&dt[(r + 8) * ED + e], 0, 3);
      __builtin_prefetch(&dbc[(r + 8) * DBCS + DTR + n], 0, 3);
    }
    const float dtv = dt[r * ED + e];
    const float xv  = xp[r * ED + e];
    const float Bm  = dbc[r * DBCS + DTR + n];
    const float dA  = expf(dtv * Aen);
    h = fmaf(dA, h, dtv * Bm * xv);
  }
  const size_t rl = rbase + (LH - 1);
  const float Cm = dbc[rl * DBCS + DTR + NN + n];
  float v = h * Cm;
  for (int m = 1; m < 16; m <<= 1) v += __shfl_xor(v, m);
  if (n == 0) {
    ypos[(size_t)branch * BB * ED + (size_t)b * ED + e] =
        v + Dp[e] * xp[rl * ED + e];
  }
}

// ---------------------------------------------------------------------------
// K7: zs = silu(xn[b,pos] @ in_proj_d[:, ED:]) for both branches
// ---------------------------------------------------------------------------
__global__ void z_kernel(const float* __restrict__ xn,
                         const float* __restrict__ wf,
                         const float* __restrict__ wb,
                         float* __restrict__ zs) {
  const int idx = blockIdx.x * blockDim.x + threadIdx.x;
  if (idx >= 2 * BB * ED) return;
  const int branch = idx / (BB * ED);
  const int rem = idx % (BB * ED);
  const int b = rem / ED, e = rem % ED;
  const float* w = branch ? wb : wf;
  const float* row = xn + ((size_t)b * LL + POS) * DD;
  float s = 0.f;
  for (int d = 0; d < DD; ++d) s += row[d] * w[(size_t)d * (2 * ED) + ED + e];
  zs[idx] = s / (1.f + expf(-s));
}

// ---------------------------------------------------------------------------
// K8: out[b] = sigmoid((cls + yd @ out_proj) @ head_w + head_b)
// ---------------------------------------------------------------------------
__global__ void final_kernel(const float* __restrict__ ypos,
                             const float* __restrict__ zs,
                             const float* __restrict__ cls,
                             const float* __restrict__ op,
                             const float* __restrict__ hw,
                             const float* __restrict__ hb,
                             float* __restrict__ out) {
  const int b = blockIdx.x;
  const int tid = threadIdx.x;
  __shared__ float yd[ED];
  __shared__ float red[256];

  for (int e = tid; e < ED; e += blockDim.x) {
    const float yf = ypos[b * ED + e] * zs[b * ED + e];
    const float yb = ypos[(BB + b) * ED + e] * zs[(BB + b) * ED + e];
    yd[e] = yf + yb;
  }
  __syncthreads();

  float partial = 0.f;
  for (int d = tid; d < DD; d += blockDim.x) {
    float o = cls[d];
    for (int e = 0; e < ED; ++e) o += yd[e] * op[(size_t)e * DD + d];
    partial += o * hw[d];
  }
  red[tid] = partial;
  __syncthreads();
  for (int s = 128; s > 0; s >>= 1) {
    if (tid < s) red[tid] += red[tid + s];
    __syncthreads();
  }
  if (tid == 0) {
    const float lg = red[0] + hb[0];
    out[b] = 1.f / (1.f + expf(-lg));
  }
}

// ---------------------------------------------------------------------------
// Launch
// ---------------------------------------------------------------------------
extern "C" void kernel_launch(void* const* d_in, const int* in_sizes, int n_in,
                              void* d_out, int out_size, void* d_ws, size_t ws_size,
                              hipStream_t stream) {
  const float* ids   = (const float*)d_in[0];
  const int*   ntw   = (const int*)d_in[1];
  const int*   nwd   = (const int*)d_in[2];
  const float* cls   = (const float*)d_in[3];
  const float* watn  = (const float*)d_in[4];
  const float* batn  = (const float*)d_in[5];
  const float* normw = (const float*)d_in[6];
  const float* ipf   = (const float*)d_in[7];
  const float* cwf   = (const float*)d_in[8];
  const float* cbf   = (const float*)d_in[9];
  const float* xprf  = (const float*)d_in[10];
  const float* dtwf  = (const float*)d_in[11];
  const float* dtbf  = (const float*)d_in[12];
  const float* alf   = (const float*)d_in[13];
  const float* dpf   = (const float*)d_in[14];
  const float* ipb   = (const float*)d_in[15];
  const float* cwb   = (const float*)d_in[16];
  const float* cbb   = (const float*)d_in[17];
  const float* xprb  = (const float*)d_in[18];
  const float* dtwb  = (const float*)d_in[19];
  const float* dtbb  = (const float*)d_in[20];
  const float* alb   = (const float*)d_in[21];
  const float* dpb   = (const float*)d_in[22];
  const float* oproj = (const float*)d_in[23];
  const float* hw    = (const float*)d_in[24];
  const float* hb    = (const float*)d_in[25];
  float* out = (float*)d_out;

  // Workspace carve-up, 256B-aligned blocks (~30 MB total; L2-resident)
  float* ws = (float*)d_ws;
  size_t o = 0;
  auto allocf = [&](size_t n) {
    float* p = ws + o;
    o += (n + 63) & ~(size_t)63;
    return p;
  };
  float* xn   = allocf((size_t)BB * LL * DD);
  float* uf   = allocf((size_t)MTOT * ED);
  float* ub   = allocf((size_t)MTOT * ED);
  float* xpf  = allocf((size_t)MTOT * ED);
  float* xpb  = allocf((size_t)MTOT * ED);
  float* dtf  = allocf((size_t)MTOT * ED);
  float* dtb  = allocf((size_t)MTOT * ED);
  float* dbcf = allocf((size_t)MTOT * DBCS);
  float* dbcb = allocf((size_t)MTOT * DBCS);
  float* zs   = allocf((size_t)2 * BB * ED);
  float* ypos = allocf((size_t)2 * BB * ED);
  // bf16 buffers (counts in bf16 elements, carved as float halves)
  __bf16* a1 = (__bf16*)allocf(((size_t)2 * MTOT * KP1 + 1) / 2);
  __bf16* a2 = (__bf16*)allocf(((size_t)2 * MTOT * KP2 + 1) / 2);
  __bf16* a3 = (__bf16*)allocf(((size_t)2 * MTOT * KP3 + 1) / 2);
  __bf16* w1 = (__bf16*)allocf(((size_t)2 * NT1 * KT1 * 512 + 1) / 2);
  __bf16* w2 = (__bf16*)allocf(((size_t)2 * NT2 * KT2 * 512 + 1) / 2);
  __bf16* w3 = (__bf16*)allocf(((size_t)2 * NT3 * KT3 * 512 + 1) / 2);

  const int MT = (MTOT + 15) / 16;  // 126

  pool_norm_kernel<<<BB * LL, 256, 0, stream>>>(ids, ntw, nwd, cls, watn, batn,
                                                normw, xn);

  // Operand packing (weights are independent of xn; A1 depends on it)
  pack_a1_kernel<<<(2 * MTOT * KP1 + 255) / 256, 256, 0, stream>>>(xn, a1);
  pack_w_kernel<<<(NT1 * KT1 * 32 + 255) / 256, 256, 0, stream>>>(
      ipf, 2 * ED, ED, DD, KT1, NT1, w1);
  pack_w_kernel<<<(NT1 * KT1 * 32 + 255) / 256, 256, 0, stream>>>(
      ipb, 2 * ED, ED, DD, KT1, NT1, w1 + (size_t)NT1 * KT1 * 512);
  pack_w_kernel<<<(NT2 * KT2 * 32 + 255) / 256, 256, 0, stream>>>(
      xprf, DTR + 2 * NN, DTR + 2 * NN, ED, KT2, NT2, w2);
  pack_w_kernel<<<(NT2 * KT2 * 32 + 255) / 256, 256, 0, stream>>>(
      xprb, DTR + 2 * NN, DTR + 2 * NN, ED, KT2, NT2,
      w2 + (size_t)NT2 * KT2 * 512);
  pack_w_kernel<<<(NT3 * KT3 * 32 + 255) / 256, 256, 0, stream>>>(
      dtwf, ED, ED, DTR, KT3, NT3, w3);
  pack_w_kernel<<<(NT3 * KT3 * 32 + 255) / 256, 256, 0, stream>>>(
      dtwb, ED, ED, DTR, KT3, NT3, w3 + (size_t)NT3 * KT3 * 512);

  inproj_kernel<<<dim3(MT, 7, 2), 128, 0, stream>>>(a1, w1, uf, ub);
  conv_silu_kernel<<<dim3((MTOT * KP2 + 255) / 256, 1, 2), 256, 0, stream>>>(
      uf, ub, cwf, cbf, cwb, cbb, xpf, xpb, a2);
  xproj_kernel<<<dim3(MT, 1, 2), 128, 0, stream>>>(a2, w2, dbcf, dbcb, a3);
  dt_kernel<<<dim3(MT, 7, 2), 128, 0, stream>>>(a3, w3, dtbf, dtbb, dtf, dtb);
  scan_kernel<<<dim3(BB * (ED / 16), 1, 2), 256, 0, stream>>>(
      xpf, xpb, dtf, dtb, dbcf, dbcb, alf, alb, dpf, dpb, ypos);
  z_kernel<<<(2 * BB * ED + 255) / 256, 256, 0, stream>>>(xn, ipf, ipb, zs);
  final_kernel<<<BB, 256, 0, stream>>>(ypos, zs, cls, oproj, hw, hb, out);
}
